// Tacotron_67250597921502
// MI455X (gfx1250) — matrix operand order repaired and checked by reading, not compiled, not run on hardware
//
#include <hip/hip_runtime.h>
#include <hip/hip_bf16.h>
#include <stddef.h>

// ---------------------------------------------------------------------------
// CDNA5 (gfx1250) Tacotron forward.
//  - All GEMMs/convs: v_wmma_f32_16x16x32_bf16, f32 accumulate.
//  - Weights pre-converted to bf16 in K-major ([N][Kp]) layout so B fragments
//    are two contiguous 16B loads (global_load_b128 / ds_load_b128).
//  - A fragments: row-major LDS tile, two contiguous 16B ds_load_b128 per lane.
//  - Recurrent (GRU) weights pinned in 320KB WGP LDS (persistent kernels).
//  - wave32; 256-thread blocks = 8 waves, decoder 512 = 16 waves.
// ---------------------------------------------------------------------------

typedef __attribute__((ext_vector_type(16))) __bf16 v16bf;
typedef __attribute__((ext_vector_type(8)))  __bf16 v8bf;
typedef __attribute__((ext_vector_type(8)))  float  v8f;

#define ACT_NONE 0
#define ACT_RELU 1
#define ACT_SIGM 2
#define ACT_TANH 3

__device__ inline __bf16 f2bf(float f) {
  union { float f; unsigned u; } in; in.f = f;
  unsigned r = (in.u + 0x7FFFu + ((in.u >> 16) & 1u)) >> 16;
  union { unsigned short s; __bf16 b; } out; out.s = (unsigned short)r; return out.b;
}
__device__ inline float b2f(__bf16 b) {
  union { unsigned short s; __bf16 b; } in; in.b = b;
  union { unsigned u; float f; } out; out.u = ((unsigned)in.s) << 16; return out.f;
}
__device__ inline float sigm(float x) { return 1.f / (1.f + __expf(-x)); }

__device__ inline v8f wmma_bf16(v16bf a, v16bf b, v8f c) {
  return __builtin_amdgcn_wmma_f32_16x16x32_bf16(false, a, false, b, (short)0, c,
                                                 false, false);
}

// A fragment (16x32 bf16, row-major, row stride ldk elements, ldk % 8 == 0).
// ISA 7.12.2: lane (m=lane%16, half=lane/16) holds
//   elems 0..7  = A[m][8*half .. 8*half+7]
//   elems 8..15 = A[m][16+8*half .. 16+8*half+7]
// -> two contiguous 16B loads (ds_load_b128 / global_load_b128).
__device__ inline v16bf load_A_frag(const __bf16* p, int ldk) {
  int lane = threadIdx.x & 31;
  int m = lane & 15, half = lane >> 4;
  const __bf16* base = p + m * ldk + 8 * half;
  v8bf lo = *(const v8bf*)(base);
  v8bf hi = *(const v8bf*)(base + 16);
  return __builtin_shufflevector(lo, hi, 0, 1, 2, 3, 4, 5, 6, 7,
                                 8, 9, 10, 11, 12, 13, 14, 15);
}

// B fragment (32x16 bf16) from K-major weights Wt[n][k] (row stride ldk).
// Lane (n=n0+lane%16, half) holds K = 8*half .. 8*half+7, then 16+8*half .. 16+8*half+7
// -> two 16B loads.
__device__ inline v16bf load_B_frag_t(const __bf16* p, int ldk, int n0) {
  int lane = threadIdx.x & 31;
  int n = n0 + (lane & 15), half = lane >> 4;
  const __bf16* base = p + (size_t)n * ldk + 8 * half;
  v8bf lo = *(const v8bf*)(base);
  v8bf hi = *(const v8bf*)(base + 16);
  return __builtin_shufflevector(lo, hi, 0, 1, 2, 3, 4, 5, 6, 7,
                                 8, 9, 10, 11, 12, 13, 14, 15);
}

// ---------------------------------------------------------------------------
// fp32 [K][N] -> bf16 K-major [N][Kp] with zero padding for k >= K
// ---------------------------------------------------------------------------
__global__ void cvt_pad_t_kernel(const float* __restrict__ src, __bf16* __restrict__ dst,
                                 int K, int N, int Kp) {
  int i = blockIdx.x * 256 + threadIdx.x;
  if (i >= N * Kp) return;
  int n = i / Kp, k = i % Kp;
  dst[i] = (k < K) ? f2bf(src[(size_t)k * N + n]) : f2bf(0.f);
}

// ---------------------------------------------------------------------------
// Generic C = act(A @ W + b), A fp32 [M][lda], Wt bf16 K-major [N][Kp],
// C fp32 [M][ldc].  Block: 256 threads (8 waves) -> 32x64 C tile.
// A tile staged in LDS; B fragments straight from global (L2-resident).
// ---------------------------------------------------------------------------
template <int ACT>
__global__ __launch_bounds__(256) void wmma_gemm_kernel(
    const float* __restrict__ A, const __bf16* __restrict__ Wt,
    const float* __restrict__ bias, float* __restrict__ C,
    int M, int N, int K, int Kp, int lda, int ldc) {
  __shared__ __bf16 As[32 * 32];
  int tid = threadIdx.x, wave = tid >> 5, wm = wave >> 2, wn = wave & 3;
  int m0 = blockIdx.x * 32, n0 = blockIdx.y * 64;
  v8f acc = {};
  for (int k0 = 0; k0 < K; k0 += 32) {
    for (int i = tid; i < 1024; i += 256) {
      int r = i >> 5, kk = i & 31;
      int m = m0 + r, k = k0 + kk;
      As[i] = f2bf((m < M && k < K) ? A[(size_t)m * lda + k] : 0.f);
    }
    if (k0 + 32 < K) __builtin_prefetch(A + (size_t)(m0 + (tid >> 3)) * lda + k0 + 32, 0, 1);
    __syncthreads();
    v16bf a = load_A_frag(As + wm * 16 * 32, 32);
    v16bf b = load_B_frag_t(Wt + k0, Kp, n0 + wn * 16);
    acc = wmma_bf16(a, b, acc);
    __syncthreads();
  }
  int lane = tid & 31, half = lane >> 4;
  int n = n0 + wn * 16 + (lane & 15);
  if (n < N) {
    float bv = bias ? bias[n] : 0.f;
#pragma unroll
    for (int v = 0; v < 8; ++v) {
      int m = m0 + wm * 16 + v + 8 * half;
      if (m < M) {
        float r = acc[v] + bv;
        if (ACT == ACT_RELU) r = fmaxf(r, 0.f);
        if (ACT == ACT_SIGM) r = sigm(r);
        if (ACT == ACT_TANH) r = tanhf(r);
        C[(size_t)m * ldc + n] = r;
      }
    }
  }
}

// ---------------------------------------------------------------------------
// 1D conv (SAME) as tap-accumulated WMMA GEMM over rows (b,t), T=512.
// Wt layout: K-major [Cout][width*Cin] (rows = output channels).
// Optional bf16 input and fused 2-wide maxpool on input; bf16 or f32 output.
// ---------------------------------------------------------------------------
template <int ACT, bool INBF, bool MAXP, bool OUTBF>
__global__ __launch_bounds__(256) void wmma_conv_kernel(
    const void* __restrict__ Xv, const __bf16* __restrict__ Wt,
    const float* __restrict__ bias, void* __restrict__ Yv,
    int Cin, int Cout, int width, int pad, int ldin, int ldout, int coloff) {
  const float*  Xf = (const float*)Xv;
  const __bf16* Xb = (const __bf16*)Xv;
  __shared__ __bf16 As[32 * 32];
  int tid = threadIdx.x, wave = tid >> 5, wm = wave >> 2, wn = wave & 3;
  int m0 = blockIdx.x * 32, n0 = blockIdx.y * 64;
  int Kt = width * Cin;  // total (padded-free) K extent, row stride of Wt
  v8f acc = {};
  for (int tap = 0; tap < width; ++tap) {
    for (int k0 = 0; k0 < Cin; k0 += 32) {
      for (int i = tid; i < 1024; i += 256) {
        int r = i >> 5, kk = i & 31;
        int m = m0 + r, b = m >> 9, t = m & 511;
        int ts = t + tap - pad;
        float v = 0.f;
        if (ts >= 0 && ts < 512) {
          int c = k0 + kk;
          if (INBF) {
            float v1 = b2f(Xb[(size_t)(b * 512 + ts) * ldin + c]);
            if (MAXP && ts + 1 < 512)
              v1 = fmaxf(v1, b2f(Xb[(size_t)(b * 512 + ts + 1) * ldin + c]));
            v = v1;
          } else {
            float v1 = Xf[(size_t)(b * 512 + ts) * ldin + c];
            if (MAXP && ts + 1 < 512)
              v1 = fmaxf(v1, Xf[(size_t)(b * 512 + ts + 1) * ldin + c]);
            v = v1;
          }
        }
        As[i] = f2bf(v);
      }
      __syncthreads();
      v16bf a = load_A_frag(As + wm * 16 * 32, 32);
      v16bf b = load_B_frag_t(Wt + tap * Cin + k0, Kt, n0 + wn * 16);
      acc = wmma_bf16(a, b, acc);
      __syncthreads();
    }
  }
  int lane = tid & 31, half = lane >> 4;
  int n = n0 + wn * 16 + (lane & 15);
  if (n < Cout) {
    float bv = bias ? bias[n] : 0.f;
#pragma unroll
    for (int v = 0; v < 8; ++v) {
      int m = m0 + wm * 16 + v + 8 * half;
      float r = acc[v] + bv;
      if (ACT == ACT_RELU) r = fmaxf(r, 0.f);
      if (OUTBF) ((__bf16*)Yv)[(size_t)m * ldout + coloff + n] = f2bf(r);
      else       ((float*)Yv)[(size_t)m * ldout + coloff + n]  = r;
    }
  }
}

// ---------------------------------------------------------------------------
// Elementwise helpers
// ---------------------------------------------------------------------------
__global__ void embed_kernel(const int* __restrict__ tok, const float* __restrict__ emb,
                             float* __restrict__ out) {
  size_t i = (size_t)blockIdx.x * 256 + threadIdx.x;
  if (i >= (size_t)16384 * 256) return;
  int row = (int)(i >> 8), c = (int)(i & 255);
  out[i] = emb[(size_t)tok[row] * 256 + c];
}
__global__ void add_kernel(float* __restrict__ x, const float* __restrict__ y, int n) {
  int i = blockIdx.x * 256 + threadIdx.x;
  if (i < n) x[i] += y[i];
}
__global__ void highway_combine_kernel(float* __restrict__ x, const float* __restrict__ Hh,
                                       const float* __restrict__ Tg, int n) {
  int i = blockIdx.x * 256 + threadIdx.x;
  if (i < n) x[i] = Hh[i] * Tg[i] + x[i] * (1.f - Tg[i]);
}

// ---------------------------------------------------------------------------
// Persistent bidirectional GRU: one block per direction, 8 waves, T=512 steps.
// Wg(256->[256][256]) + Wc([128][256]) bf16 K-major pinned in LDS.
// Writes memory[b][t][dir*128 + c] directly (B=32,T=512,256).
// ---------------------------------------------------------------------------
__global__ __launch_bounds__(256) void gru_kernel(
    const float* __restrict__ x,
    const __bf16* Wg_f, const float* bg_f, const __bf16* Wc_f, const float* bc_f,
    const __bf16* Wg_b, const float* bg_b, const __bf16* Wc_b, const float* bc_b,
    float* __restrict__ mem) {
  extern __shared__ char sm[];
  __bf16* Wg  = (__bf16*)sm;          // [256][256] K-major (N rows, K cols)
  __bf16* Wc  = Wg + 256 * 256;       // [128][256] K-major
  __bf16* cat = Wc + 128 * 256;       // [32][256]
  float*  h     = (float*)(cat + 32 * 256);  // 32*128
  float*  gates = h + 32 * 128;              // 32*256
  float*  cand  = gates + 32 * 256;          // 32*128
  int tid = threadIdx.x, dir = blockIdx.x;
  const __bf16* Wgg = dir ? Wg_b : Wg_f;
  const __bf16* Wcg = dir ? Wc_b : Wc_f;
  const float*  bgp = dir ? bg_b : bg_f;
  const float*  bcp = dir ? bc_b : bc_f;
  for (int i = tid; i < 256 * 256; i += 256) Wg[i] = Wgg[i];
  for (int i = tid; i < 128 * 256; i += 256) Wc[i] = Wcg[i];
  for (int i = tid; i < 32 * 128; i += 256) h[i] = 0.f;
  __syncthreads();
  int wave = tid >> 5, lane = tid & 31, half = lane >> 4;
  for (int s = 0; s < 512; ++s) {
    int t = dir ? 511 - s : s;
    for (int i = tid; i < 4096; i += 256) {
      int b = i >> 7, c = i & 127;
      cat[b * 256 + c]       = f2bf(x[(size_t)(b * 512 + t) * 128 + c]);
      cat[b * 256 + 128 + c] = f2bf(h[i]);
    }
    __syncthreads();
    for (int tile = wave; tile < 32; tile += 8) {   // gates: 32x256
      int wm = tile & 1, wn = tile >> 1;
      v8f acc = {};
#pragma unroll
      for (int kc = 0; kc < 256; kc += 32) {
        v16bf a = load_A_frag(cat + wm * 16 * 256 + kc, 256);
        v16bf b = load_B_frag_t(Wg + kc, 256, wn * 16);
        acc = wmma_bf16(a, b, acc);
      }
      int n = wn * 16 + (lane & 15);
      float bv = bgp[n];
#pragma unroll
      for (int v = 0; v < 8; ++v) {
        int m = wm * 16 + v + 8 * half;
        gates[m * 256 + n] = sigm(acc[v] + bv);
      }
    }
    __syncthreads();
    for (int i = tid; i < 4096; i += 256) {         // cat[:,128:] = r*h
      int b = i >> 7, c = i & 127;
      cat[b * 256 + 128 + c] = f2bf(gates[b * 256 + c] * h[i]);
    }
    __syncthreads();
    for (int tile = wave; tile < 16; tile += 8) {   // candidate: 32x128
      int wm = tile & 1, wn = tile >> 1;
      v8f acc = {};
#pragma unroll
      for (int kc = 0; kc < 256; kc += 32) {
        v16bf a = load_A_frag(cat + wm * 16 * 256 + kc, 256);
        v16bf b = load_B_frag_t(Wc + kc, 256, wn * 16);
        acc = wmma_bf16(a, b, acc);
      }
      int n = wn * 16 + (lane & 15);
      float bv = bcp[n];
#pragma unroll
      for (int v = 0; v < 8; ++v) {
        int m = wm * 16 + v + 8 * half;
        cand[m * 128 + n] = tanhf(acc[v] + bv);
      }
    }
    __syncthreads();
    for (int i = tid; i < 4096; i += 256) {
      int b = i >> 7, c = i & 127;
      float u = gates[b * 256 + 128 + c];
      float hn = u * h[i] + (1.f - u) * cand[i];
      h[i] = hn;
      mem[(size_t)(b * 512 + t) * 256 + dir * 128 + c] = hn;
    }
    __syncthreads();
  }
}

// ---------------------------------------------------------------------------
// Decoder: one persistent block, 16 waves, 500 sequential steps.
// ---------------------------------------------------------------------------
struct DecParams {
  const float* outputs;  // (32,1000,81) teacher source
  const float* keys;     // (32,512,256)
  const float* memory;   // (32,512,256)
  const __bf16 *W0, *W1, *Wq, *Wg_att, *Wc_att, *Wg_d0, *Wc_d0, *Wg_d1, *Wc_d1, *Wmgc;
  const float  *b0, *b1, *bg_att, *bc_att, *bg_d0, *bc_d0, *bg_d1, *bc_d1, *bmgc;
  const float  *att_v, *Wvuv, *bvuv;
  float* out_final;  // (1000,32,81)
  float* out_alpha;  // (32,500,512)
};

// 16-wave cooperative GEMM: out[32][N] = act(A(lds bf16, [32][Kp]) @ Wt + b),
// Wt K-major [N][Kp] in global.
__device__ void block_gemm16(int tid, const __bf16* Abuf, int Kp,
                             const __bf16* Wt, int N, const float* bias,
                             float* out, int act) {
  int wave = tid >> 5, lane = tid & 31, half = lane >> 4;
  int tiles = 2 * (N >> 4);
  for (int tile = wave; tile < tiles; tile += 16) {
    int wm = tile & 1, wn = tile >> 1;
    v8f acc = {};
    for (int k0 = 0; k0 < Kp; k0 += 32) {
      v16bf a = load_A_frag(Abuf + wm * 16 * Kp + k0, Kp);
      v16bf b = load_B_frag_t(Wt + k0, Kp, wn * 16);
      acc = wmma_bf16(a, b, acc);
    }
    int n = wn * 16 + (lane & 15);
    float bv = bias ? bias[n] : 0.f;
#pragma unroll
    for (int v = 0; v < 8; ++v) {
      int m = wm * 16 + v + 8 * half;
      float r = acc[v] + bv;
      if (act == ACT_RELU) r = fmaxf(r, 0.f);
      else if (act == ACT_SIGM) r = sigm(r);
      else if (act == ACT_TANH) r = tanhf(r);
      out[m * N + n] = r;
    }
  }
}

// GRU step with H=256. Abuf[:, :Kin] pre-staged with x (stride Kin+256).
__device__ void dec_gru(int tid, __bf16* Abuf, int Kin, float* h,
                        const __bf16* Wgm, const float* bg,
                        const __bf16* Wcm, const float* bc,
                        float* gates, float* cand) {
  int Ktot = Kin + 256;
  for (int i = tid; i < 32 * 256; i += 512) {
    int b = i >> 8, c = i & 255;
    Abuf[b * Ktot + Kin + c] = f2bf(h[i]);
  }
  __syncthreads();
  block_gemm16(tid, Abuf, Ktot, Wgm, 512, bg, gates, ACT_SIGM);
  __syncthreads();
  for (int i = tid; i < 32 * 256; i += 512) {
    int b = i >> 8, c = i & 255;
    Abuf[b * Ktot + Kin + c] = f2bf(gates[b * 512 + c] * h[i]);
  }
  __syncthreads();
  block_gemm16(tid, Abuf, Ktot, Wcm, 256, bc, cand, ACT_TANH);
  __syncthreads();
  for (int i = tid; i < 32 * 256; i += 512) {
    int b = i >> 8, c = i & 255;
    float u = gates[b * 512 + 256 + c];
    h[i] = u * h[i] + (1.f - u) * cand[i];
  }
  __syncthreads();
}

__global__ __launch_bounds__(512) void decoder_kernel(DecParams P) {
  extern __shared__ char sm[];
  __bf16* Abuf = (__bf16*)sm;                       // 32*768
  float* gates = (float*)(sm + 32 * 768 * 2);       // 32*512
  float* tmp1  = gates + 32 * 512;                  // 32*256
  float* tmp2  = tmp1 + 32 * 256;                   // 32*256
  float* h_att = tmp2 + 32 * 256;                   // 32*256
  float* hd0   = h_att + 32 * 256;                  // 32*256
  float* hd1   = hd0 + 32 * 256;                    // 32*256
  float* vvec  = hd1 + 32 * 256;                    // 256
  int tid = threadIdx.x;
  for (int i = tid; i < 32 * 256; i += 512) { h_att[i] = 0.f; hd0[i] = 0.f; hd1[i] = 0.f; }
  if (tid < 256) vvec[tid] = P.att_v[tid];
  __syncthreads();

  for (int step = 0; step < 500; ++step) {
    // ---- dec_pre: q0 = relu(x @ W0 + b0), x = teacher[step] (81 -> pad 96)
    for (int i = tid; i < 32 * 96; i += 512) {
      int b = i / 96, j = i % 96;
      float v = 0.f;
      if (step > 0 && j < 81) v = P.outputs[(size_t)(b * 1000 + 2 * step - 2) * 81 + j];
      Abuf[i] = f2bf(v);
    }
    __syncthreads();
    block_gemm16(tid, Abuf, 96, P.W0, 256, P.b0, tmp1, ACT_RELU);
    __syncthreads();
    for (int i = tid; i < 32 * 256; i += 512) Abuf[i] = f2bf(tmp1[i]);
    __syncthreads();
    block_gemm16(tid, Abuf, 256, P.W1, 128, P.b1, tmp2, ACT_RELU);  // q [32][128]
    __syncthreads();
    // ---- attention RNN (Kin=128, Ktot=384)
    for (int i = tid; i < 32 * 128; i += 512) {
      int b = i >> 7, c = i & 127;
      Abuf[b * 384 + c] = f2bf(tmp2[b * 128 + c]);
    }
    dec_gru(tid, Abuf, 128, h_att, P.Wg_att, P.bg_att, P.Wc_att, P.bc_att, gates, tmp1);
    // ---- qk = h_att @ att_Wq
    for (int i = tid; i < 32 * 256; i += 512) Abuf[i] = f2bf(h_att[i]);
    __syncthreads();
    block_gemm16(tid, Abuf, 256, P.Wq, 256, nullptr, tmp1, ACT_NONE);
    __syncthreads();
    // ---- scores e[b][t] = sum_d tanh(keys + qk) * v
    for (int p = tid; p < 32 * 512; p += 512) {
      int b = p >> 9, t = p & 511;
      const float* kr = P.keys + (size_t)(b * 512 + t) * 256;
      const float* qr = tmp1 + b * 256;
      float s = 0.f;
      for (int d = 0; d < 256; ++d) s += tanhf(kr[d] + qr[d]) * vvec[d];
      gates[p] = s;
    }
    __syncthreads();
    // ---- softmax over t (one wave per batch row), write alpha output
    {
      int wave = tid >> 5, lane = tid & 31;
      for (int b = wave; b < 32; b += 16) {
        float* e = gates + b * 512;
        float mx = -1e30f;
        for (int t = lane; t < 512; t += 32) mx = fmaxf(mx, e[t]);
        for (int off = 16; off; off >>= 1) mx = fmaxf(mx, __shfl_xor(mx, off, 32));
        float sum = 0.f;
        for (int t = lane; t < 512; t += 32) { float ex = __expf(e[t] - mx); e[t] = ex; sum += ex; }
        for (int off = 16; off; off >>= 1) sum += __shfl_xor(sum, off, 32);
        float inv = 1.f / sum;
        for (int t = lane; t < 512; t += 32) {
          float a = e[t] * inv;
          e[t] = a;
          P.out_alpha[(size_t)(b * 500 + step) * 512 + t] = a;
        }
      }
    }
    __syncthreads();
    // ---- ctx[b][d] = sum_t alpha * memory
    for (int i = tid; i < 32 * 256; i += 512) {
      int b = i >> 8, d = i & 255;
      const float* mb = P.memory + (size_t)b * 512 * 256 + d;
      const float* ab = gates + b * 512;
      float s = 0.f;
      for (int t = 0; t < 512; ++t) s += ab[t] * mb[(size_t)t * 256];
      tmp2[i] = s;
    }
    __syncthreads();
    // ---- dec0: x=[ctx, h_att] (Kin=512, Ktot=768)
    for (int i = tid; i < 32 * 256; i += 512) {
      int b = i >> 8, c = i & 255;
      Abuf[b * 768 + c]       = f2bf(tmp2[i]);
      Abuf[b * 768 + 256 + c] = f2bf(h_att[i]);
    }
    dec_gru(tid, Abuf, 512, hd0, P.Wg_d0, P.bg_d0, P.Wc_d0, P.bc_d0, gates, tmp1);
    // ---- dec1: x=hd0 (Kin=256, Ktot=512)
    for (int i = tid; i < 32 * 256; i += 512) {
      int b = i >> 8, c = i & 255;
      Abuf[b * 512 + c] = f2bf(hd0[i]);
    }
    dec_gru(tid, Abuf, 256, hd1, P.Wg_d1, P.bg_d1, P.Wc_d1, P.bc_d1, gates, tmp1);
    // ---- o = hd1 + hd0
    for (int i = tid; i < 32 * 256; i += 512) {
      float o = hd1[i] + hd0[i];
      tmp1[i] = o;
      Abuf[i] = f2bf(o);
    }
    __syncthreads();
    // ---- mgc = o @ W_mgc + b (N=160), into gates buffer
    block_gemm16(tid, Abuf, 256, P.Wmgc, 160, P.bmgc, gates, ACT_NONE);
    __syncthreads();
    // ---- vuv (N=2, VALU) + writeback into final (1000,32,81)
    if (tid < 64) {
      int b = tid >> 1, r = tid & 1;
      const float* o = tmp1 + b * 256;
      float s = P.bvuv[r];
      for (int d = 0; d < 256; ++d) s += o[d] * P.Wvuv[d * 2 + r];
      P.out_final[((size_t)(2 * step + r) * 32 + b) * 81 + 80] = sigm(s);
    }
    for (int i = tid; i < 32 * 160; i += 512) {
      int b = i / 160, rem = i % 160, r = rem / 80, j = rem % 80;
      P.out_final[((size_t)(2 * step + r) * 32 + b) * 81 + j] = gates[b * 160 + rem];
    }
    __syncthreads();
  }
}

// ---------------------------------------------------------------------------
// Host orchestration
// ---------------------------------------------------------------------------
extern "C" void kernel_launch(void* const* d_in, const int* in_sizes, int n_in,
                              void* d_out, int out_size, void* d_ws, size_t ws_size,
                              hipStream_t stream) {
  (void)in_sizes; (void)n_in; (void)out_size; (void)ws_size;
  const int M = 32 * 512;  // 16384 rows

  // ---- input pointers (setup_inputs dict order, params depth-first) ----
  const int*   tokens  = (const int*)d_in[0];
  const float* outputs = (const float*)d_in[1];
  const float* embed   = (const float*)d_in[2];
  const float* W0e = (const float*)d_in[3];  const float* b0e = (const float*)d_in[4];
  const float* W1e = (const float*)d_in[5];  const float* b1e = (const float*)d_in[6];
  const float *bank_w[16], *bank_b[16];
  for (int k = 0; k < 16; ++k) { bank_w[k] = (const float*)d_in[7 + 2 * k]; bank_b[k] = (const float*)d_in[8 + 2 * k]; }
  const float* proj1_w = (const float*)d_in[39]; const float* proj1_b = (const float*)d_in[40];
  const float* proj2_w = (const float*)d_in[41]; const float* proj2_b = (const float*)d_in[42];
  const float *hw_Wh[4], *hw_bh[4], *hw_Wt[4], *hw_bt[4];
  for (int l = 0; l < 4; ++l) {
    hw_Wh[l] = (const float*)d_in[43 + 4 * l]; hw_bh[l] = (const float*)d_in[44 + 4 * l];
    hw_Wt[l] = (const float*)d_in[45 + 4 * l]; hw_bt[l] = (const float*)d_in[46 + 4 * l];
  }
  const float* gfw_Wg = (const float*)d_in[59]; const float* gfw_bg = (const float*)d_in[60];
  const float* gfw_Wc = (const float*)d_in[61]; const float* gfw_bc = (const float*)d_in[62];
  const float* gbw_Wg = (const float*)d_in[63]; const float* gbw_bg = (const float*)d_in[64];
  const float* gbw_Wc = (const float*)d_in[65]; const float* gbw_bc = (const float*)d_in[66];
  const float* att_Wm = (const float*)d_in[67];
  const float* att_Wq = (const float*)d_in[68];
  const float* att_v  = (const float*)d_in[69];
  const float* dW0 = (const float*)d_in[70]; const float* db0 = (const float*)d_in[71];
  const float* dW1 = (const float*)d_in[72]; const float* db1 = (const float*)d_in[73];
  const float* ar_Wg = (const float*)d_in[74]; const float* ar_bg = (const float*)d_in[75];
  const float* ar_Wc = (const float*)d_in[76]; const float* ar_bc = (const float*)d_in[77];
  const float* d0_Wg = (const float*)d_in[78]; const float* d0_bg = (const float*)d_in[79];
  const float* d0_Wc = (const float*)d_in[80]; const float* d0_bc = (const float*)d_in[81];
  const float* d1_Wg = (const float*)d_in[82]; const float* d1_bg = (const float*)d_in[83];
  const float* d1_Wc = (const float*)d_in[84]; const float* d1_bc = (const float*)d_in[85];
  const float* W_mgc = (const float*)d_in[86]; const float* b_mgc = (const float*)d_in[87];
  const float* W_vuv = (const float*)d_in[88]; const float* b_vuv = (const float*)d_in[89];

  // ---- workspace carve-out ----
  size_t cur = 0;
  auto alloc = [&](size_t bytes) -> void* {
    void* p = (char*)d_ws + cur;
    cur = (cur + bytes + 255) & ~(size_t)255;
    return p;
  };
  // fp32 [K][N] -> bf16 K-major [N][Kp] (zero-padded K)
  auto cvt = [&](const float* src, int K, int N, int Kp) -> __bf16* {
    __bf16* dst = (__bf16*)alloc((size_t)Kp * N * sizeof(__bf16));
    int n = Kp * N;
    cvt_pad_t_kernel<<<(n + 255) / 256, 256, 0, stream>>>(src, dst, K, N, Kp);
    return dst;
  };

  // bf16 K-major weight pool
  __bf16* W0e_h = cvt(W0e, 256, 256, 256);
  __bf16* W1e_h = cvt(W1e, 256, 128, 256);
  __bf16* bankW_h[16];
  for (int k = 0; k < 16; ++k) bankW_h[k] = cvt(bank_w[k], (k + 1) * 128, 128, (k + 1) * 128);
  __bf16* p1W_h = cvt(proj1_w, 3 * 2048, 128, 3 * 2048);
  __bf16* p2W_h = cvt(proj2_w, 3 * 128, 128, 3 * 128);
  __bf16 *hwWh_h[4], *hwWt_h[4];
  for (int l = 0; l < 4; ++l) { hwWh_h[l] = cvt(hw_Wh[l], 128, 128, 128); hwWt_h[l] = cvt(hw_Wt[l], 128, 128, 128); }
  __bf16* gfwWg_h = cvt(gfw_Wg, 256, 256, 256);
  __bf16* gfwWc_h = cvt(gfw_Wc, 256, 128, 256);
  __bf16* gbwWg_h = cvt(gbw_Wg, 256, 256, 256);
  __bf16* gbwWc_h = cvt(gbw_Wc, 256, 128, 256);
  __bf16* attWm_h = cvt(att_Wm, 256, 256, 256);
  __bf16* attWq_h = cvt(att_Wq, 256, 256, 256);
  __bf16* dW0_h   = cvt(dW0, 81, 256, 96);     // pad K 81 -> 96
  __bf16* dW1_h   = cvt(dW1, 256, 128, 256);
  __bf16* arWg_h  = cvt(ar_Wg, 384, 512, 384);
  __bf16* arWc_h  = cvt(ar_Wc, 384, 256, 384);
  __bf16* d0Wg_h  = cvt(d0_Wg, 768, 512, 768);
  __bf16* d0Wc_h  = cvt(d0_Wc, 768, 256, 768);
  __bf16* d1Wg_h  = cvt(d1_Wg, 512, 512, 512);
  __bf16* d1Wc_h  = cvt(d1_Wc, 512, 256, 512);
  __bf16* Wmgc_h  = cvt(W_mgc, 256, 160, 256);

  // activation buffers
  float*  emb_t  = (float*)alloc((size_t)M * 256 * 4);
  float*  h_t    = (float*)alloc((size_t)M * 256 * 4);
  float*  pre_t  = (float*)alloc((size_t)M * 128 * 4);
  __bf16* bank_t = (__bf16*)alloc((size_t)M * 2048 * 2);
  float*  pr_t   = (float*)alloc((size_t)M * 128 * 4);
  float*  x_t    = (float*)alloc((size_t)M * 128 * 4);
  float*  mem_t  = (float*)alloc((size_t)M * 256 * 4);
  float*  keys_t = (float*)alloc((size_t)M * 256 * 4);

  const size_t GRU_SMEM = (size_t)(256 * 256 + 128 * 256 + 32 * 256) * 2 +
                          (size_t)(32 * 128 + 32 * 256 + 32 * 128) * 4;
  const size_t DEC_SMEM = (size_t)32 * 768 * 2 + (size_t)(32 * 512 + 5 * 32 * 256 + 256) * 4;
  (void)hipFuncSetAttribute((const void*)gru_kernel,
                            hipFuncAttributeMaxDynamicSharedMemorySize, (int)GRU_SMEM);
  (void)hipFuncSetAttribute((const void*)decoder_kernel,
                            hipFuncAttributeMaxDynamicSharedMemorySize, (int)DEC_SMEM);

  // ---- encoder ----
  {
    size_t n = (size_t)M * 256;
    embed_kernel<<<(unsigned)((n + 255) / 256), 256, 0, stream>>>(tokens, embed, emb_t);
  }
  wmma_gemm_kernel<ACT_RELU><<<dim3(M / 32, 4), 256, 0, stream>>>(emb_t, W0e_h, b0e, h_t, M, 256, 256, 256, 256, 256);
  wmma_gemm_kernel<ACT_RELU><<<dim3(M / 32, 2), 256, 0, stream>>>(h_t, W1e_h, b1e, pre_t, M, 128, 256, 256, 256, 128);
  for (int k = 0; k < 16; ++k) {
    int w = k + 1, pad = k / 2;
    wmma_conv_kernel<ACT_RELU, false, false, true><<<dim3(M / 32, 2), 256, 0, stream>>>(
        pre_t, bankW_h[k], bank_b[k], bank_t, 128, 128, w, pad, 128, 2048, k * 128);
  }
  wmma_conv_kernel<ACT_RELU, true, true, false><<<dim3(M / 32, 2), 256, 0, stream>>>(
      bank_t, p1W_h, proj1_b, pr_t, 2048, 128, 3, 1, 2048, 128, 0);
  wmma_conv_kernel<ACT_NONE, false, false, false><<<dim3(M / 32, 2), 256, 0, stream>>>(
      pr_t, p2W_h, proj2_b, x_t, 128, 128, 3, 1, 128, 128, 0);
  {
    int n = M * 128;
    add_kernel<<<(n + 255) / 256, 256, 0, stream>>>(x_t, pre_t, n);
  }
  for (int l = 0; l < 4; ++l) {
    wmma_gemm_kernel<ACT_RELU><<<dim3(M / 32, 2), 256, 0, stream>>>(x_t, hwWh_h[l], hw_bh[l], emb_t, M, 128, 128, 128, 128, 128);
    wmma_gemm_kernel<ACT_SIGM><<<dim3(M / 32, 2), 256, 0, stream>>>(x_t, hwWt_h[l], hw_bt[l], h_t, M, 128, 128, 128, 128, 128);
    int n = M * 128;
    highway_combine_kernel<<<(n + 255) / 256, 256, 0, stream>>>(x_t, emb_t, h_t, n);
  }
  // ---- bidirectional GRU (weights pinned in LDS) ----
  gru_kernel<<<2, 256, (unsigned)GRU_SMEM, stream>>>(
      x_t, gfwWg_h, gfw_bg, gfwWc_h, gfw_bc, gbwWg_h, gbw_bg, gbwWc_h, gbw_bc, mem_t);
  // ---- attention keys ----
  wmma_gemm_kernel<ACT_NONE><<<dim3(M / 32, 4), 256, 0, stream>>>(mem_t, attWm_h, nullptr, keys_t, M, 256, 256, 256, 256, 256);
  // ---- decoder ----
  DecParams P;
  P.outputs = outputs; P.keys = keys_t; P.memory = mem_t;
  P.W0 = dW0_h; P.W1 = dW1_h; P.Wq = attWq_h;
  P.Wg_att = arWg_h; P.Wc_att = arWc_h;
  P.Wg_d0 = d0Wg_h; P.Wc_d0 = d0Wc_h;
  P.Wg_d1 = d1Wg_h; P.Wc_d1 = d1Wc_h;
  P.Wmgc = Wmgc_h;
  P.b0 = db0; P.b1 = db1; P.bg_att = ar_bg; P.bc_att = ar_bc;
  P.bg_d0 = d0_bg; P.bc_d0 = d0_bc; P.bg_d1 = d1_bg; P.bc_d1 = d1_bc;
  P.bmgc = b_mgc; P.att_v = att_v; P.Wvuv = W_vuv; P.bvuv = b_vuv;
  P.out_final = (float*)d_out;
  P.out_alpha = (float*)d_out + (size_t)1000 * 32 * 81;
  decoder_kernel<<<1, 512, (unsigned)DEC_SMEM, stream>>>(P);
}
